// CoherenceGuidedAttention_81922206204609
// MI455X (gfx1250) — matrix operand order repaired and checked by reading, not compiled, not run on hardware
//
#include <hip/hip_runtime.h>
#include <hip/hip_bf16.h>
#include <math.h>
#include <stdint.h>

typedef __attribute__((ext_vector_type(16))) _Float16 v16h;
typedef __attribute__((ext_vector_type(2)))  _Float16 v2h;
typedef __attribute__((ext_vector_type(8)))  float    v8f;

#define B_    8
#define C_    128
#define N_    1024
#define HEADS 8
#define HD    16

// workspace byte offsets
#define OFF_W    0                      // w:    B*N f32      = 32 KB
#define OFF_GATE 32768                  // gate: B f32
#define OFF_Q    36864                  // Qh:   B*N*C f16    = 2 MB
#define OFF_K    (OFF_Q + 2097152)      // Kh
#define OFF_V    (OFF_K + 2097152)      // Vh
#define OFF_O    (OFF_V + 2097152)      // Oh (per-head attn out, f16)
// total = OFF_O + 2 MB = 8,425,472 bytes

// ---------------------------------------------------------------------------
// CDNA5 async global->LDS copy (16 bytes per lane) + ASYNCcnt waits.
// The builtin's pointee type is a GCC vector 'int __attribute__((vector_size(16)))'
// in addrspace(1)/(3); cast via integers (low 32 bits of a generic LDS VA are
// the LDS byte offset per the ISA aperture rules).
// ---------------------------------------------------------------------------
typedef int i128v __attribute__((vector_size(16)));
typedef __attribute__((address_space(1))) i128v as1_i128;
typedef __attribute__((address_space(3))) i128v as3_i128;

__device__ __forceinline__ void async_copy16(const void* g, void* l) {
#if __has_builtin(__builtin_amdgcn_global_load_async_to_lds_b128)
  __builtin_amdgcn_global_load_async_to_lds_b128(
      (as1_i128*)(uintptr_t)g,
      (as3_i128*)(unsigned int)(uintptr_t)l,
      0, 0);
#else
  unsigned int       loff = (unsigned int)(uintptr_t)l;
  unsigned long long ga   = (unsigned long long)(uintptr_t)g;
  asm volatile("global_load_async_to_lds_b128 %0, %1, off"
               :: "v"(loff), "v"(ga) : "memory");
#endif
}

template <int N>
__device__ __forceinline__ void wait_async() {
#if __has_builtin(__builtin_amdgcn_s_wait_asynccnt)
  __builtin_amdgcn_s_wait_asynccnt(N);
#else
  if constexpr (N == 0) asm volatile("s_wait_asynccnt 0x0" ::: "memory");
  else                  asm volatile("s_wait_asynccnt 0x1" ::: "memory");
#endif
}

// ---------------------------------------------------------------------------
// Kernel 1: coherence weights + stats + pooled features + gate MLP
// ---------------------------------------------------------------------------
__global__ __launch_bounds__(256)
void cga_gate_kernel(const float* __restrict__ vv, const float* __restrict__ vh,
                     const float* __restrict__ coh,
                     const float* __restrict__ g1w, const float* __restrict__ g1b,
                     const float* __restrict__ g2w, const float* __restrict__ g2b,
                     const float* __restrict__ g3w, const float* __restrict__ g3b,
                     float* __restrict__ w_out, float* __restrict__ gate_out) {
  const int b = blockIdx.x, t = threadIdx.x;
  __shared__ float r0[256], r1[256];
  __shared__ float gi[262];
  __shared__ float h1[64], h2[32];
  __shared__ float s_min, s_inv;

  const float* cb = coh + b * N_;
  float v0 = cb[t], v1 = cb[t+256], v2 = cb[t+512], v3 = cb[t+768];
  float lmin = fminf(fminf(v0, v1), fminf(v2, v3));
  float lmax = fmaxf(fmaxf(v0, v1), fmaxf(v2, v3));
  r0[t] = lmin; r1[t] = lmax; __syncthreads();
  for (int s = 128; s > 0; s >>= 1) {
    if (t < s) { r0[t] = fminf(r0[t], r0[t+s]); r1[t] = fmaxf(r1[t], r1[t+s]); }
    __syncthreads();
  }
  if (t == 0) { s_min = r0[0]; s_inv = 1.0f / (r1[0] - r0[0] + 1e-8f); }
  __syncthreads();
  const float cmin = s_min, inv = s_inv;
  float w0 = (v0-cmin)*inv, w1 = (v1-cmin)*inv, w2 = (v2-cmin)*inv, w3 = (v3-cmin)*inv;
  float* wb = w_out + b * N_;
  wb[t] = w0; wb[t+256] = w1; wb[t+512] = w2; wb[t+768] = w3;

  r0[t] = w0 + w1 + w2 + w3;
  r1[t] = w0*w0 + w1*w1 + w2*w2 + w3*w3;
  __syncthreads();
  for (int s = 128; s > 0; s >>= 1) {
    if (t < s) { r0[t] += r0[t+s]; r1[t] += r1[t+s]; }
    __syncthreads();
  }
  if (t == 0) {
    float mean = r0[0] * (1.0f / N_);
    float var  = r1[0] * (1.0f / N_) - mean * mean;
    gi[256] = mean; gi[257] = sqrtf(fmaxf(var, 0.0f));
  }
  __syncthreads();
  r0[t] = fmaxf(fmaxf(w0, w1), fmaxf(w2, w3));
  __syncthreads();
  for (int s = 128; s > 0; s >>= 1) {
    if (t < s) r0[t] = fmaxf(r0[t], r0[t+s]);
    __syncthreads();
  }
  if (t == 0) gi[258] = r0[0];

  // channel-pooled means: one warp (32 lanes) per channel
  const int warp = t >> 5, lane = t & 31;
  for (int it = 0; it < 16; ++it) {
    int c = it * 8 + warp;
    const float* pv = vv + (size_t)b*C_*N_ + (size_t)c*N_;
    const float* ph = vh + (size_t)b*C_*N_ + (size_t)c*N_;
    float sv = 0.f, sh = 0.f;
    for (int j = lane; j < N_; j += 32) { sv += pv[j]; sh += ph[j]; }
    for (int m = 16; m > 0; m >>= 1) { sv += __shfl_xor(sv, m, 32); sh += __shfl_xor(sh, m, 32); }
    if (lane == 0) { gi[c] = sv * (1.0f / N_); gi[C_ + c] = sh * (1.0f / N_); }
  }
  __syncthreads();
  if (t < 64) {
    float a = g1b[t];
    for (int i = 0; i < 259; ++i) a += gi[i] * g1w[t*259 + i];
    h1[t] = fmaxf(a, 0.0f);
  }
  __syncthreads();
  if (t < 32) {
    float a = g2b[t];
    for (int i = 0; i < 64; ++i) a += h1[i] * g2w[t*64 + i];
    h2[t] = fmaxf(a, 0.0f);
  }
  __syncthreads();
  if (t == 0) {
    float a = g3b[0];
    for (int i = 0; i < 32; ++i) a += h2[i] * g3w[i];
    gate_out[b] = 1.0f / (1.0f + __expf(-a));
  }
}

// ---------------------------------------------------------------------------
// Kernel 2: Q/K/V projections via WMMA.  grid (128, 3): y selects Q/K/V.
// Each block: 64 tokens x 128 out-channels, 8 waves in 4(M)x2(N) layout.
// (A needs a transpose, B needs f32->f16 conversion, so staging stays sync.)
// ---------------------------------------------------------------------------
__global__ __launch_bounds__(256)
void cga_qkv_kernel(const float* __restrict__ vv, const float* __restrict__ vh,
                    const float* __restrict__ Wq, const float* __restrict__ bq,
                    const float* __restrict__ Wk, const float* __restrict__ bk,
                    const float* __restrict__ Wv, const float* __restrict__ bv,
                    _Float16* __restrict__ Qh, _Float16* __restrict__ Kh,
                    _Float16* __restrict__ Vh) {
  const int which = blockIdx.y;
  const float* src = (which == 0) ? vv : vh;
  const float* Wm  = (which == 0) ? Wq : (which == 1) ? Wk : Wv;
  const float* bi  = (which == 0) ? bq : (which == 1) ? bk : bv;
  _Float16*    dst = (which == 0) ? Qh : (which == 1) ? Kh : Vh;

  const int t   = threadIdx.x;
  const int mt  = blockIdx.x;        // 0..127
  const int b   = mt >> 4;
  const int n0  = (mt & 15) * 64;    // token base within batch
  const int m0g = mt * 64;           // flat token base

  __shared__ _Float16 Ah[64 * 130];  // tokens x channels (pad -> conflict-free)
  __shared__ _Float16 Bh[128 * 130]; // out-ch x in-ch

  #pragma unroll
  for (int i = 0; i < 32; ++i) {
    int idx = t + i * 256;
    int m = idx & 63, k = idx >> 6;
    Ah[m*130 + k] = (_Float16)src[(size_t)b*C_*N_ + (size_t)k*N_ + n0 + m];
  }
  #pragma unroll
  for (int i = 0; i < 32; ++i) {
    int idx = t + i * 256;
    int k2 = (idx & 63) * 2, n = idx >> 6;
    float2 f = *(const float2*)(Wm + n*C_ + k2);
    v2h p; p[0] = (_Float16)f.x; p[1] = (_Float16)f.y;
    *(v2h*)(Bh + n*130 + k2) = p;
  }
  __syncthreads();

  const int w  = t >> 5, lane = t & 31;
  const int wm = w & 3,  wn   = w >> 2;
  const int hi = lane >> 4, lc = lane & 15;

  v8f acc[4];
  #pragma unroll
  for (int j = 0; j < 4; ++j) {
    float bval = bi[wn*64 + j*16 + lc];
    #pragma unroll
    for (int r = 0; r < 8; ++r) acc[j][r] = bval;
  }
  #pragma unroll
  for (int kk = 0; kk < 128; kk += 32) {
    v16h af;
    const _Float16* arow = Ah + (wm*16 + lc) * 130;
    #pragma unroll
    for (int v = 0; v < 8; ++v) {
      int k = kk + ((v >= 4) ? 16 : 0) + (v & 3)*2 + hi*8;
      v2h p = *(const v2h*)(arow + k);
      af[2*v] = p[0]; af[2*v+1] = p[1];
    }
    #pragma unroll
    for (int j = 0; j < 4; ++j) {
      v16h bf;
      const _Float16* brow = Bh + (wn*64 + j*16 + lc) * 130;
      #pragma unroll
      for (int v = 0; v < 8; ++v) {
        int k = kk + ((v >= 4) ? 16 : 0) + (v & 3)*2 + hi*8;
        v2h p = *(const v2h*)(brow + k);
        bf[2*v] = p[0]; bf[2*v+1] = p[1];
      }
      acc[j] = __builtin_amdgcn_wmma_f32_16x16x32_f16(false, af, false, bf,
                                                      (short)0, acc[j], false, false);
    }
  }
  #pragma unroll
  for (int j = 0; j < 4; ++j) {
    int n = wn*64 + j*16 + lc;
    #pragma unroll
    for (int r = 0; r < 8; ++r) {
      int m = wm*16 + r + hi*8;
      dst[(size_t)(m0g + m)*C_ + n] = (_Float16)acc[j][r];
    }
  }
}

// ---------------------------------------------------------------------------
// Kernel 3: fused attention per (batch, head, 64-query tile).
// Full 64x1024 f32 score stripe in the 320 KB WGP LDS.  K tiles are staged
// with double-buffered GLOBAL_LOAD_ASYNC_TO_LDS_B128 (ASYNCcnt waits).
// ---------------------------------------------------------------------------
#define KS_STRIDE 24   // halfs; 48 B rows: 16-B aligned, conflict-free gather

__global__ __launch_bounds__(256)
void cga_attn_kernel(const _Float16* __restrict__ Qh, const _Float16* __restrict__ Kh,
                     const _Float16* __restrict__ Vh, const float* __restrict__ w_buf,
                     const float* __restrict__ gate_buf, const float* __restrict__ temp,
                     _Float16* __restrict__ Oh) {
  const int mtile = blockIdx.x;   // 0..15
  const int h     = blockIdx.y;   // 0..7
  const int b     = blockIdx.z;   // 0..7
  const int t     = threadIdx.x;
  const int m0    = mtile * 64;

  __shared__ float    Ssh[64 * 1028];          // 263168 B, padded stride
  __shared__ _Float16 Vt[16 * 1032];           // V transposed [d][n], padded
  __shared__ _Float16 Qs[64 * 18];             // scaled Q tile
  __shared__ _Float16 Ks[2][64 * KS_STRIDE];   // double-buffered K tiles
  __shared__ float    wkAll[1024];
  __shared__ float    wq[64];
  __shared__ float    rstat[256];
  __shared__ float    rowmax[64], rowsum[64];

  const float gate   = gate_buf[b];
  const float qscale = temp[0] * gate;
  const float ginv   = 1.0f - gate;

  // async K-tile stager: 64 rows x 32 B, two B128 chunks per row
  auto stageK = [&](int kt, _Float16* buf) {
    if (t < 128) {
      int j = t >> 1, hf = t & 1;
      async_copy16(Kh + (size_t)(b*N_ + kt*64 + j)*C_ + h*HD + hf*8,
                   buf + j*KS_STRIDE + hf*8);
    }
  };
  stageK(0, Ks[0]);   // prefetch first tile behind phase-0 staging

  #pragma unroll
  for (int i = 0; i < 2; ++i) {               // stage scaled Q (64x16 halfs)
    int idx = t + i * 256;
    int m = idx >> 3, k2 = (idx & 7) * 2;
    v2h p = *(const v2h*)(Qh + (size_t)(b*N_ + m0 + m)*C_ + h*HD + k2);
    v2h q; q[0] = (_Float16)((float)p[0] * qscale);
           q[1] = (_Float16)((float)p[1] * qscale);
    *(v2h*)(Qs + m*18 + k2) = q;
  }
  #pragma unroll 4
  for (int i = 0; i < 64; ++i) {              // stage V transposed
    int idx = t + i * 256;
    int d = idx & 15, n = idx >> 4;
    Vt[d*1032 + n] = Vh[(size_t)(b*N_ + n)*C_ + h*HD + d];
  }
  if (t < 64) wq[t] = ginv * w_buf[b*N_ + m0 + t];
  #pragma unroll
  for (int i = 0; i < 4; ++i) wkAll[t + i*256] = w_buf[b*N_ + t + i*256];
  __syncthreads();

  const int w  = t >> 5, lane = t & 31;
  const int hi = lane >> 4, lc = lane & 15;

  // ---- phase 1: scores (double-buffered K tiles) ----
  for (int kt = 0; kt < 16; ++kt) {
    const _Float16* cur = Ks[kt & 1];
    if (kt < 15) { stageK(kt + 1, Ks[(kt + 1) & 1]); wait_async<1>(); }
    else         { wait_async<0>(); }
    __syncthreads();

    const int nn = w >> 1;
    v16h bf;
    #pragma unroll
    for (int v = 0; v < 8; ++v) { bf[2*v] = (_Float16)0; bf[2*v+1] = (_Float16)0; }
    {                                         // K dim is 16: elements 8..15 zero
      const _Float16* krow = cur + (nn*16 + lc) * KS_STRIDE;
      #pragma unroll
      for (int v = 0; v < 4; ++v) {
        v2h p = *(const v2h*)(krow + 2*v + hi*8);
        bf[2*v] = p[0]; bf[2*v+1] = p[1];
      }
    }
    #pragma unroll
    for (int wmi = 0; wmi < 2; ++wmi) {
      const int wm = (w & 1)*2 + wmi;
      v16h af;
      #pragma unroll
      for (int v = 0; v < 8; ++v) { af[2*v] = (_Float16)0; af[2*v+1] = (_Float16)0; }
      const _Float16* qrow = Qs + (wm*16 + lc) * 18;
      #pragma unroll
      for (int v = 0; v < 4; ++v) {
        v2h p = *(const v2h*)(qrow + v*2 + hi*8);
        af[2*v] = p[0]; af[2*v+1] = p[1];
      }
      v8f acc = {};
      acc = __builtin_amdgcn_wmma_f32_16x16x32_f16(false, af, false, bf,
                                                   (short)0, acc, false, false);
      int ncol = kt*64 + nn*16 + lc;
      float wkv = wkAll[ncol];
      #pragma unroll
      for (int r = 0; r < 8; ++r) {
        int mrow = wm*16 + r + hi*8;
        Ssh[mrow*1028 + ncol] = acc[r] + wq[mrow] * wkv;
      }
    }
    __syncthreads();
  }

  // ---- phase 2: softmax in place (normalization deferred) ----
  {
    int row = t >> 2, part = t & 3;
    float* sp = Ssh + row*1028 + part*256;
    float mx = -3.0e38f;
    for (int i = 0; i < 256; ++i) mx = fmaxf(mx, sp[i]);
    rstat[t] = mx;
    __syncthreads();
    if (t < 64)
      rowmax[t] = fmaxf(fmaxf(rstat[t*4], rstat[t*4+1]),
                        fmaxf(rstat[t*4+2], rstat[t*4+3]));
    __syncthreads();
    float rm = rowmax[row], sm = 0.f;
    for (int i = 0; i < 256; ++i) { float e = __expf(sp[i] - rm); sp[i] = e; sm += e; }
    rstat[t] = sm;
    __syncthreads();
    if (t < 64) rowsum[t] = rstat[t*4] + rstat[t*4+1] + rstat[t*4+2] + rstat[t*4+3];
    __syncthreads();
  }

  // ---- phase 3: P @ V ----
  {
    const int wm = w & 3, hsel = w >> 2;     // each output tile split over 2 waves
    v8f acc = {};
    for (int t8 = 0; t8 < 8; ++t8) {
      int nb0 = hsel*512 + t8*64;
      #pragma unroll
      for (int kk = 0; kk < 64; kk += 32) {
        int nb = nb0 + kk;
        v16h af;
        const float* srow = Ssh + (wm*16 + lc) * 1028 + nb;
        #pragma unroll
        for (int v = 0; v < 8; ++v) {
          int k = ((v >= 4) ? 16 : 0) + (v & 3)*2 + hi*8;
          float2 p = *(const float2*)(srow + k);
          af[2*v] = (_Float16)p.x; af[2*v+1] = (_Float16)p.y;
        }
        v16h bf;
        const _Float16* vrow = Vt + lc*1032 + nb;
        #pragma unroll
        for (int v = 0; v < 8; ++v) {
          v2h p = *(const v2h*)(vrow + ((v >= 4) ? 16 : 0) + (v & 3)*2 + hi*8);
          bf[2*v] = p[0]; bf[2*v+1] = p[1];
        }
        acc = __builtin_amdgcn_wmma_f32_16x16x32_f16(false, af, false, bf,
                                                     (short)0, acc, false, false);
      }
    }
    __syncthreads();                 // all S reads done; safe to reuse as scratch
    float* comb = Ssh;
    if (hsel == 1) {
      #pragma unroll
      for (int r = 0; r < 8; ++r)
        comb[wm*256 + (r + hi*8)*16 + lc] = acc[r];
    }
    __syncthreads();
    if (hsel == 0) {
      #pragma unroll
      for (int r = 0; r < 8; ++r) {
        int mrow = wm*16 + r + hi*8;
        float o = (acc[r] + comb[wm*256 + (r + hi*8)*16 + lc]) / rowsum[mrow];
        Oh[(size_t)(b*N_ + m0 + mrow)*C_ + h*HD + lc] = (_Float16)o;
      }
    }
  }
}

// ---------------------------------------------------------------------------
// Kernel 4: out = vv + (Oh @ Wo^T + bo), transposed back to [B,C,H,W].
// A tile (f16, contiguous rows) is staged with async global->LDS B128 copies
// while the f32->f16 W conversion proceeds on the VALU.
// ---------------------------------------------------------------------------
#define AH_STRIDE 136  // halfs; 272 B rows: 16-B aligned, conflict-free gather

__global__ __launch_bounds__(256)
void cga_outproj_kernel(const _Float16* __restrict__ Oh, const float* __restrict__ Wo,
                        const float* __restrict__ bo, const float* __restrict__ vv,
                        float* __restrict__ out) {
  const int t = threadIdx.x;
  const int m0g = blockIdx.x * 64;
  __shared__ _Float16 Ah[64 * AH_STRIDE];
  __shared__ _Float16 Bh[128 * 130];
  __shared__ float    Ot[128 * 65];

  // async stage A: 64 rows x 256 B = 1024 B128 chunks
  #pragma unroll
  for (int i = 0; i < 4; ++i) {
    int idx = t + i * 256;
    int m = idx >> 4, c16 = idx & 15;
    async_copy16(Oh + (size_t)(m0g + m)*C_ + c16*8, Ah + m*AH_STRIDE + c16*8);
  }
  // meanwhile: stage + convert W on the VALU
  #pragma unroll
  for (int i = 0; i < 32; ++i) {
    int idx = t + i * 256;
    int k2 = (idx & 63) * 2, n = idx >> 6;
    float2 f = *(const float2*)(Wo + n*C_ + k2);
    v2h p; p[0] = (_Float16)f.x; p[1] = (_Float16)f.y;
    *(v2h*)(Bh + n*130 + k2) = p;
  }
  wait_async<0>();
  __syncthreads();

  const int w  = t >> 5, lane = t & 31;
  const int wm = w & 3,  wn   = w >> 2;
  const int hi = lane >> 4, lc = lane & 15;

  v8f acc[4];
  #pragma unroll
  for (int j = 0; j < 4; ++j) {
    float bval = bo[wn*64 + j*16 + lc];
    #pragma unroll
    for (int r = 0; r < 8; ++r) acc[j][r] = bval;
  }
  #pragma unroll
  for (int kk = 0; kk < 128; kk += 32) {
    v16h af;
    const _Float16* arow = Ah + (wm*16 + lc) * AH_STRIDE;
    #pragma unroll
    for (int v = 0; v < 8; ++v) {
      int k = kk + ((v >= 4) ? 16 : 0) + (v & 3)*2 + hi*8;
      v2h p = *(const v2h*)(arow + k);
      af[2*v] = p[0]; af[2*v+1] = p[1];
    }
    #pragma unroll
    for (int j = 0; j < 4; ++j) {
      v16h bf;
      const _Float16* brow = Bh + (wn*64 + j*16 + lc) * 130;
      #pragma unroll
      for (int v = 0; v < 8; ++v) {
        int k = kk + ((v >= 4) ? 16 : 0) + (v & 3)*2 + hi*8;
        v2h p = *(const v2h*)(brow + k);
        bf[2*v] = p[0]; bf[2*v+1] = p[1];
      }
      acc[j] = __builtin_amdgcn_wmma_f32_16x16x32_f16(false, af, false, bf,
                                                      (short)0, acc[j], false, false);
    }
  }
  #pragma unroll
  for (int j = 0; j < 4; ++j) {
    int n = wn*64 + j*16 + lc;
    #pragma unroll
    for (int r = 0; r < 8; ++r) {
      int m = wm*16 + r + hi*8;
      Ot[n*65 + m] = acc[j][r];
    }
  }
  __syncthreads();
  #pragma unroll
  for (int i = 0; i < 32; ++i) {
    int idx = t + i * 256;
    int c = idx >> 6, mm = idx & 63;
    int gm = m0g + mm, b = gm >> 10, tok = gm & 1023;
    size_t o = (size_t)b*C_*N_ + (size_t)c*N_ + tok;
    out[o] = vv[o] + Ot[c*65 + mm];
  }
}

// ---------------------------------------------------------------------------
extern "C" void kernel_launch(void* const* d_in, const int* in_sizes, int n_in,
                              void* d_out, int out_size, void* d_ws, size_t ws_size,
                              hipStream_t stream) {
  const float* vv   = (const float*)d_in[0];
  const float* vh   = (const float*)d_in[1];
  const float* coh  = (const float*)d_in[2];
  const float* Wq   = (const float*)d_in[3];
  const float* bq   = (const float*)d_in[4];
  const float* Wk   = (const float*)d_in[5];
  const float* bk   = (const float*)d_in[6];
  const float* Wv   = (const float*)d_in[7];
  const float* bv   = (const float*)d_in[8];
  const float* Wo   = (const float*)d_in[9];
  const float* bo   = (const float*)d_in[10];
  const float* temp = (const float*)d_in[11];
  const float* g1w  = (const float*)d_in[12];
  const float* g1b  = (const float*)d_in[13];
  const float* g2w  = (const float*)d_in[14];
  const float* g2b  = (const float*)d_in[15];
  const float* g3w  = (const float*)d_in[16];
  const float* g3b  = (const float*)d_in[17];
  float* out = (float*)d_out;

  char* ws = (char*)d_ws;
  float*    w_buf    = (float*)(ws + OFF_W);
  float*    gate_buf = (float*)(ws + OFF_GATE);
  _Float16* Qh       = (_Float16*)(ws + OFF_Q);
  _Float16* Kh       = (_Float16*)(ws + OFF_K);
  _Float16* Vh       = (_Float16*)(ws + OFF_V);
  _Float16* Oh       = (_Float16*)(ws + OFF_O);

  cga_gate_kernel<<<B_, 256, 0, stream>>>(vv, vh, coh, g1w, g1b, g2w, g2b,
                                          g3w, g3b, w_buf, gate_buf);
  cga_qkv_kernel<<<dim3(128, 3), 256, 0, stream>>>(vv, vh, Wq, bq, Wk, bk,
                                                   Wv, bv, Qh, Kh, Vh);
  cga_attn_kernel<<<dim3(16, HEADS, B_), 256, 0, stream>>>(Qh, Kh, Vh, w_buf,
                                                           gate_buf, temp, Oh);
  cga_outproj_kernel<<<128, 256, 0, stream>>>(Oh, Wo, bo, vv, out);
}
